// MfaPtnetSsn_70952859730207
// MI455X (gfx1250) — matrix operand order repaired and checked by reading, not compiled, not run on hardware
//
#include <hip/hip_runtime.h>
#include <hip/hip_bf16.h>

// ---------------------------------------------------------------------------
// CDNA5 (gfx1250): WMMA f16 GEMMs with fragment-ready LDS tiles (ds_load_b128
// operand fetch), f16 global activations, async global->LDS staging
// (global_load_async_to_lds_b128 + s_wait_asynccnt) for GEMM input tiles.
// ---------------------------------------------------------------------------

typedef __attribute__((ext_vector_type(16))) _Float16 v16h;
typedef __attribute__((ext_vector_type(8)))  _Float16 v8h;
typedef __attribute__((ext_vector_type(8)))  float    v8f;

#define Bn 4
#define Npts 4096
#define NSPIX 128
#define FD 64
#define KNN_K 32
#define KSLIC 16

__device__ __forceinline__ v8f wmma32(v16h a, v16h b, v8f c) {
  return __builtin_amdgcn_wmma_f32_16x16x32_f16(false, a, false, b,
                                                (short)0, c, false, false);
}

// Fragment-ready tile: 512 halfs = 32 lanes x 16 contiguous halfs.
// A tile (16Mx32K): lane<16: row=lane, halfs i -> K i;
//                   lane>=16: row=lane-16, halfs i -> K 16+i.
// B tile (32Kx16N): lane<16: col=lane, halfs i -> K i; lane>=16: col=lane-16, K 16+i.
__device__ __forceinline__ v16h frag_ld(const _Float16* tileBase) {
  int lane = threadIdx.x & 31;
  return *(const v16h*)(tileBase + lane * 16);
}

// Stage A-frag tiles (mt-major, then kt) from f32 row-major W (Cout x Cin).
__device__ __forceinline__ void stage_a_frag(_Float16* dst, const float* W,
                                             int Cout, int Cin, int CoutP, int CinP) {
  int mtiles = CoutP >> 4, ktiles = CinP >> 5;
  for (int t = threadIdx.x; t < mtiles * ktiles * 32; t += blockDim.x) {
    int lane = t & 31, tile = t >> 5;
    int mt = tile / ktiles, kt = tile - mt * ktiles;
    int row = mt * 16 + (lane & 15);
    int kbase = kt * 32 + ((lane & 16) ? 16 : 0);
    v8h lo, hi;
#pragma unroll
    for (int i = 0; i < 8; ++i) {
      int c0 = kbase + i, c1 = kbase + 8 + i;
      lo[i] = (_Float16)((row < Cout && c0 < Cin) ? W[row * Cin + c0] : 0.f);
      hi[i] = (_Float16)((row < Cout && c1 < Cin) ? W[row * Cin + c1] : 0.f);
    }
    _Float16* p = dst + tile * 512 + lane * 16;
    *(v8h*)p = lo;
    *(v8h*)(p + 8) = hi;
  }
}

// read pf[p][c] back out of an A-frag layout (inverse map)
__device__ __forceinline__ float afrag_read(const _Float16* base, int p, int c) {
  int pt = p >> 4, row = p & 15;
  int kt = c >> 5, kk = c & 31;
  int lane = row + ((kk & 16) ? 16 : 0);
  int ii = kk & 15;
  return (float)base[(pt * 2 + kt) * 512 + lane * 16 + ii];
}

// ---------------------------------------------------------------------------
// KNN: one wave per query point; full distance row in LDS, 32 argmin rounds.
// ---------------------------------------------------------------------------
__global__ void knn_kernel(const float* __restrict__ x, int* __restrict__ knn_idx,
                           float* __restrict__ knn_d2) {
  extern __shared__ float sDist[];              // 8 waves * 4096 floats
  int tid = threadIdx.x, wave = tid >> 5, lane = tid & 31;
  long gq = (long)blockIdx.x * 8 + wave;
  int b = (int)(gq / Npts), n = (int)(gq % Npts);
  const float* px = x + (long)b * 3 * Npts;
  float qx = px[n], qy = px[Npts + n], qz = px[2 * Npts + n];
  float* dw = sDist + wave * Npts;
  for (int j = lane; j < Npts; j += 32) {
    float dx = px[j] - qx, dy = px[Npts + j] - qy, dz = px[2 * Npts + j] - qz;
    dw[j] = dx * dx + dy * dy + dz * dz;
  }
  __syncthreads();
  for (int r = 0; r < KNN_K; ++r) {
    float vmin = 3.0e38f; int vidx = Npts;
    for (int t = 0; t < Npts / 32; ++t) {
      int j = t * 32 + lane;
      float d = dw[j];
      if (d < vmin || (d == vmin && j < vidx)) { vmin = d; vidx = j; }
    }
#pragma unroll
    for (int off = 16; off > 0; off >>= 1) {
      float ov = __shfl_xor(vmin, off);
      int   oi = __shfl_xor(vidx, off);
      if (ov < vmin || (ov == vmin && oi < vidx)) { vmin = ov; vidx = oi; }
    }
    if (lane == 0) {
      knn_idx[gq * KNN_K + r] = vidx;
      knn_d2 [gq * KNN_K + r] = vmin;
      dw[vidx] = 3.0e38f;
    }
    __syncthreads();
  }
}

// ---------------------------------------------------------------------------
// Branch kernel: 16 points (512 group elems) per WG. Layer1 (3->32) VALU into
// B-frag LDS (packed b128 stores); layers 2/3 WMMA with frag-ready operands;
// ReLU fused; max-pool over K via monotone int atomicMax.
// ---------------------------------------------------------------------------
__global__ void branch_kernel(const float* __restrict__ x,
                              const int* __restrict__ knn_idx,
                              const float* __restrict__ knn_d2,
                              const float* __restrict__ W1, const float* __restrict__ b1,
                              const float* __restrict__ W2, const float* __restrict__ b2,
                              const float* __restrict__ W3, const float* __restrict__ b3,
                              _Float16* __restrict__ feats, float r2, int chOff) {
  extern __shared__ char smemB[];
  _Float16* sH1  = (_Float16*)smemB;            // 32 B-frag tiles (nt)        16384 halfs
  _Float16* sH2  = sH1 + 32 * 512;              // 64 B-frag tiles (kt*32+nt)  32768 halfs
  _Float16* sW2f = sH2 + 64 * 512;              // 4 A-frag tiles (mt)          2048 halfs
  _Float16* sW3f = sW2f + 4 * 512;              // 8 A-frag tiles (mt*2+kt)     4096 halfs
  int*      sMax = (int*)(sW3f + 8 * 512);      // 64 x 16
  int b = blockIdx.y, n0 = blockIdx.x * 16, tid = threadIdx.x;

  stage_a_frag(sW2f, W2, 64, 32, 64, 32);
  stage_a_frag(sW3f, W3, 64, 64, 64, 64);
  for (int i = tid; i < 64 * 16; i += 256) sMax[i] = 0;

  const float* px = x + (long)b * 3 * Npts;
  for (int col = tid; col < 512; col += 256) {
    int p = col >> 5, kk = col & 31, n = n0 + p;
    long gi = ((long)b * Npts + n) * KNN_K + kk;
    int nbr = (knn_d2[gi] <= r2) ? knn_idx[gi] : n;
    float rx = px[nbr]            - px[n];
    float ry = px[Npts + nbr]     - px[Npts + n];
    float rz = px[2 * Npts + nbr] - px[2 * Npts + n];
    float hv[32];
#pragma unroll
    for (int o = 0; o < 32; ++o)
      hv[o] = fmaxf(W1[o * 3] * rx + W1[o * 3 + 1] * ry + W1[o * 3 + 2] * rz + b1[o], 0.f);
    v8h p0, p1, p2, p3;
#pragma unroll
    for (int i = 0; i < 8; ++i) {
      p0[i] = (_Float16)hv[i];      p1[i] = (_Float16)hv[8 + i];
      p2[i] = (_Float16)hv[16 + i]; p3[i] = (_Float16)hv[24 + i];
    }
    int nt = col >> 4, lb = col & 15;
    _Float16* base = sH1 + nt * 512;
    *(v8h*)(base + lb * 16)            = p0;  // K 0-7   (lane lb)
    *(v8h*)(base + lb * 16 + 8)        = p1;  // K 8-15
    *(v8h*)(base + (lb + 16) * 16)     = p2;  // K 16-23 (lane lb+16)
    *(v8h*)(base + (lb + 16) * 16 + 8) = p3;  // K 24-31
  }
  __syncthreads();

  int wave = tid >> 5, lane = tid & 31;
  v8f zero = {};
  // layer 2: 64x512, K=32 -> (4 x 32) tiles
  for (int t = wave; t < 128; t += 8) {
    int mt = t >> 5, nt = t & 31;
    v8f acc = wmma32(frag_ld(sW2f + mt * 512), frag_ld(sH1 + nt * 512), zero);
    int r0 = (lane & 16) ? 8 : 0;
    int rowbase = mt * 16 + r0;
    int kt = rowbase >> 5;
    int lane_b = (lane & 15) + ((rowbase & 16) ? 16 : 0);
    int ii = rowbase & 15;
    v8h pk;
#pragma unroll
    for (int r = 0; r < 8; ++r)
      pk[r] = (_Float16)fmaxf(acc[r] + b2[rowbase + r], 0.f);
    *(v8h*)(sH2 + (kt * 32 + nt) * 512 + lane_b * 16 + ii) = pk;
  }
  __syncthreads();
  // layer 3: K=64, fused ReLU + maxpool over the 32-neighbor groups
  for (int t = wave; t < 128; t += 8) {
    int mt = t >> 5, nt = t & 31;
    v8f acc = zero;
#pragma unroll
    for (int kt = 0; kt < 2; ++kt)
      acc = wmma32(frag_ld(sW3f + (mt * 2 + kt) * 512), frag_ld(sH2 + (kt * 32 + nt) * 512), acc);
    int col = nt * 16 + (lane & 15);
    int p = col >> 5;
    int r0 = (lane & 16) ? 8 : 0;
#pragma unroll
    for (int r = 0; r < 8; ++r) {
      int row = mt * 16 + r0 + r;
      float v = fmaxf(acc[r] + b3[row], 0.f);
      atomicMax(&sMax[row * 16 + p], __float_as_int(v));
    }
  }
  __syncthreads();
  for (int i = tid; i < 64 * 16; i += 256) {
    int c = i >> 4, p = i & 15;
    feats[((long)b * 192 + chOff + c) * Npts + n0 + p] = (_Float16)__int_as_float(sMax[i]);
  }
}

// ---------------------------------------------------------------------------
// Generic pointwise-MLP GEMM: Y = act(W @ X + bias), 128-point column tile.
// f16 X tiles staged with global_load_async_to_lds_b128 (raw rows), then
// shuffled LDS->LDS into B-frag tiles; f32 X (first ptnet layer) staged direct.
// ---------------------------------------------------------------------------
__global__ void mlp_gemm(const void* __restrict__ X, int xIsF16, long xbs,
                         const float* __restrict__ Wg, const float* __restrict__ bias,
                         void* __restrict__ Y, int yIsF32, long ybs,
                         int Cin, int Cout, int relu) {
  extern __shared__ char smemG[];
  int CinP = (Cin + 31) & ~31;
  int CoutP = (Cout + 15) & ~15;
  int ktiles = CinP >> 5, mtiles = CoutP >> 4;
  _Float16* sWf = (_Float16*)smemG;             // A-frag: mtiles*ktiles tiles
  _Float16* sXf = sWf + CoutP * CinP;           // B-frag: ktiles*8 tiles
  _Float16* sStage = sXf + CinP * 128;          // raw rows: CinP x 128 halfs
  unsigned stageOff = (unsigned)((CoutP * CinP + CinP * 128) * 2); // LDS byte off of sStage
  int b = blockIdx.y, n0 = blockIdx.x * 128, tid = threadIdx.x;

  stage_a_frag(sWf, Wg, Cout, Cin, CoutP, CinP);

  if (xIsF16) {
    // async raw copy of Cin rows x 256B into sStage
    const _Float16* Xh = (const _Float16*)X + (long)b * xbs;
    for (int t = tid; t < Cin * 16; t += 256) {
      int row = t >> 4, ch = t & 15;
      unsigned voff  = (unsigned)(((long)row * Npts + n0) * 2 + ch * 16);
      unsigned laddr = stageOff + (unsigned)(row * 256 + ch * 16);
      asm volatile("global_load_async_to_lds_b128 %0, %1, %2"
                   :: "v"(laddr), "v"(voff), "s"((unsigned long long)Xh) : "memory");
    }
    asm volatile("s_wait_asynccnt 0x0" ::: "memory");
    __syncthreads();
    // shuffle rows -> B-frag tiles
    for (int t = tid; t < ktiles * 8 * 32; t += 256) {
      int lane = t & 31, tile = t >> 5;
      int kt = tile >> 3, nt = tile & 7;
      int col = nt * 16 + (lane & 15);
      int kb = kt * 32 + ((lane & 16) ? 16 : 0);
      v8h lo, hi;
#pragma unroll
      for (int i = 0; i < 8; ++i) {
        lo[i] = sStage[(kb + i) * 128 + col];
        hi[i] = sStage[(kb + 8 + i) * 128 + col];
      }
      _Float16* p = sXf + tile * 512 + lane * 16;
      *(v8h*)p = lo;
      *(v8h*)(p + 8) = hi;
    }
  } else {
    const float* Xf = (const float*)X + (long)b * xbs;
    for (int t = tid; t < ktiles * 8 * 32; t += 256) {
      int lane = t & 31, tile = t >> 5;
      int kt = tile >> 3, nt = tile & 7;
      int col = n0 + nt * 16 + (lane & 15);
      int kb = kt * 32 + ((lane & 16) ? 16 : 0);
      v8h lo, hi;
#pragma unroll
      for (int i = 0; i < 8; ++i) {
        int c0 = kb + i, c1 = kb + 8 + i;
        lo[i] = (_Float16)((c0 < Cin) ? Xf[(long)c0 * Npts + col] : 0.f);
        hi[i] = (_Float16)((c1 < Cin) ? Xf[(long)c1 * Npts + col] : 0.f);
      }
      _Float16* p = sXf + tile * 512 + lane * 16;
      *(v8h*)p = lo;
      *(v8h*)(p + 8) = hi;
    }
  }
  __syncthreads();

  int wave = tid >> 5, lane = tid & 31;
  v8f zero = {};
  for (int t = wave; t < mtiles * 8; t += 8) {
    int mt = t >> 3, nt = t & 7;
    v8f acc = zero;
    for (int kt = 0; kt < ktiles; ++kt)
      acc = wmma32(frag_ld(sWf + (mt * ktiles + kt) * 512),
                   frag_ld(sXf + (kt * 8 + nt) * 512), acc);
    int col = nt * 16 + (lane & 15);
    int r0 = (lane & 16) ? 8 : 0;
#pragma unroll
    for (int r = 0; r < 8; ++r) {
      int row = mt * 16 + r0 + r;
      if (row < Cout) {
        float v = acc[r] + bias[row];
        if (relu) v = fmaxf(v, 0.f);
        if (yIsF32) ((float*)Y + (long)b * ybs)[(long)row * Npts + n0 + col] = v;
        else ((_Float16*)Y + (long)b * ybs)[(long)row * Npts + n0 + col] = (_Float16)v;
      }
    }
  }
}

// ---------------------------------------------------------------------------
// Soft-SLIC assignment: 128 points/WG (16 per wave), distances via WMMA.
// mode 0: accumulate num/w (LDS atomics, then global atomics).
// mode 1: write sparse Q rows + hard labels.
// ---------------------------------------------------------------------------
__global__ void slic_assign(const float* __restrict__ pf,
                            const float* __restrict__ centers,
                            float* __restrict__ num, float* __restrict__ wsum,
                            float* __restrict__ Qout, float* __restrict__ hardOut,
                            int mode) {
  extern __shared__ char smemS[];
  _Float16* sCf  = (_Float16*)smemS;                 // 16 B-frag tiles (kt*8+nt)
  float*    sCsq = (float*)(sCf + 16 * 512);         // 128
  _Float16* sPFf = (_Float16*)(sCsq + 128);          // 16 A-frag tiles (pt*2+kt)
  float*    sD   = (float*)(sPFf + 16 * 512);        // 8 waves x 16 x 128
  float*    sNum = sD + 8 * 16 * 128;                // 64 x 128
  float*    sW   = sNum + 64 * 128;                  // 128
  int*      sIdx = (int*)(sW + 128);                 // 128 x 16
  float*    sAff = (float*)(sIdx + 128 * 16);        // 128 x 16
  int b = blockIdx.y, n0 = blockIdx.x * 128, tid = threadIdx.x;
  const float* pfb = pf + (long)b * FD * Npts;
  const float* cb  = centers + (long)b * FD * NSPIX;

  for (int i = tid; i < 64 * 128; i += 256) sNum[i] = 0.f;
  if (tid < 128) {
    float s = 0.f;
    for (int k = 0; k < 64; ++k) { float v = cb[k * 128 + tid]; s += v * v; }
    sCsq[tid] = s;
    sW[tid] = 0.f;
  }
  // centers -> B-frag tiles
  for (int t = tid; t < 16 * 32; t += 256) {
    int lane = t & 31, tile = t >> 5;
    int kt = tile >> 3, nt = tile & 7;
    int s = nt * 16 + (lane & 15);
    int kb = kt * 32 + ((lane & 16) ? 16 : 0);
    v8h lo, hi;
#pragma unroll
    for (int i = 0; i < 8; ++i) {
      lo[i] = (_Float16)cb[(kb + i) * 128 + s];
      hi[i] = (_Float16)cb[(kb + 8 + i) * 128 + s];
    }
    _Float16* p = sCf + tile * 512 + lane * 16;
    *(v8h*)p = lo;
    *(v8h*)(p + 8) = hi;
  }
  // point features -> A-frag tiles
  for (int t = tid; t < 16 * 32; t += 256) {
    int lane = t & 31, tile = t >> 5;
    int pt = tile >> 1, kt = tile & 1;
    int p = pt * 16 + (lane & 15);
    int kbase = kt * 32 + ((lane & 16) ? 16 : 0);
    v8h lo, hi;
#pragma unroll
    for (int i = 0; i < 8; ++i) {
      lo[i] = (_Float16)pfb[(long)(kbase + i) * Npts + n0 + p];
      hi[i] = (_Float16)pfb[(long)(kbase + 8 + i) * Npts + n0 + p];
    }
    _Float16* q = sPFf + tile * 512 + lane * 16;
    *(v8h*)q = lo;
    *(v8h*)(q + 8) = hi;
  }
  __syncthreads();

  int wave = tid >> 5, lane = tid & 31;
  float* dW = sD + wave * 16 * 128;
  v8f zero = {};
  for (int nt = 0; nt < 8; ++nt) {
    v8f acc = zero;
#pragma unroll
    for (int kt = 0; kt < 2; ++kt)
      acc = wmma32(frag_ld(sPFf + (wave * 2 + kt) * 512),
                   frag_ld(sCf + (kt * 8 + nt) * 512), acc);
    int col = nt * 16 + (lane & 15);
    int r0 = (lane & 16) ? 8 : 0;
#pragma unroll
    for (int r = 0; r < 8; ++r)
      dW[(r0 + r) * 128 + col] = sCsq[col] - 2.f * acc[r];
  }
  __syncthreads();

  if (lane < 16) {                       // one lane per point: top-16 + softmax
    int p = wave * 16 + lane;
    float* dr = dW + lane * 128;
    float vals[KSLIC]; int idxs[KSLIC];
#pragma unroll 1
    for (int sel = 0; sel < KSLIC; ++sel) {
      float vmin = 3.0e38f; int vi = 0;
      for (int s = 0; s < NSPIX; ++s) { float d = dr[s]; if (d < vmin) { vmin = d; vi = s; } }
      vals[sel] = vmin; idxs[sel] = vi; dr[vi] = 3.0e38f;
    }
    float d0 = vals[0], sum = 0.f, e[KSLIC];
#pragma unroll
    for (int i = 0; i < KSLIC; ++i) { e[i] = __expf(d0 - vals[i]); sum += e[i]; }
    float inv = 1.f / sum;
#pragma unroll
    for (int i = 0; i < KSLIC; ++i) { sIdx[p * 16 + i] = idxs[i]; sAff[p * 16 + i] = e[i] * inv; }
  }
  __syncthreads();

  if (mode == 0) {
    for (int ePair = tid; ePair < 128 * 16; ePair += 256) {
      int p = ePair >> 4, s = sIdx[ePair];
      float a = sAff[ePair];
      atomicAdd(&sW[s], a);
#pragma unroll 1
      for (int c = 0; c < 64; ++c)
        atomicAdd(&sNum[c * 128 + s], a * afrag_read(sPFf, p, c));
    }
    __syncthreads();
    for (int i = tid; i < 64 * 128; i += 256)
      atomicAdd(&num[(long)b * FD * NSPIX + i], sNum[i]);
    if (tid < 128) atomicAdd(&wsum[b * NSPIX + tid], sW[tid]);
  } else {
    for (int ePair = tid; ePair < 128 * 16; ePair += 256) {
      int p = ePair >> 4, i = ePair & 15, n = n0 + p, s = sIdx[ePair];
      atomicAdd(&Qout[((long)b * NSPIX + s) * Npts + n], sAff[ePair]);
      if (i == 0) hardOut[(long)b * Npts + n] = (float)s;
    }
  }
}

__global__ void center_init(const float* __restrict__ net, float* __restrict__ centers) {
  int i = blockIdx.x * 256 + threadIdx.x;           // B*64*128
  if (i >= Bn * FD * NSPIX) return;
  int b = i / (FD * NSPIX), rem = i % (FD * NSPIX);
  int c = rem / NSPIX, s = rem % NSPIX;
  centers[i] = net[((long)b * FD + c) * Npts + s];
}

__global__ void center_update(const float* __restrict__ num, const float* __restrict__ wsum,
                              float* __restrict__ centers) {
  int i = blockIdx.x * 256 + threadIdx.x;
  if (i >= Bn * FD * NSPIX) return;
  int b = i / (FD * NSPIX), s = i % NSPIX;
  centers[i] = num[i] / (wsum[b * NSPIX + s] + 1e-8f);
}

__global__ void zero_kernel(float* __restrict__ p, long n) {
  long i = (long)blockIdx.x * 256 + threadIdx.x;
  for (; i < n; i += (long)gridDim.x * 256) p[i] = 0.f;
}

__global__ void copy_kernel(float* __restrict__ dst, const float* __restrict__ src, long n) {
  long i = (long)blockIdx.x * 256 + threadIdx.x;
  for (; i < n; i += (long)gridDim.x * 256) dst[i] = src[i];
}

// ---------------------------------------------------------------------------
extern "C" void kernel_launch(void* const* d_in, const int* in_sizes, int n_in,
                              void* d_out, int out_size, void* d_ws, size_t ws_size,
                              hipStream_t stream) {
  (void)in_sizes; (void)n_in; (void)out_size; (void)ws_size;
  const float* x = (const float*)d_in[0];
  const float *brW[3][3], *brB[3][3];
  int p = 1;
  for (int br = 0; br < 3; ++br)
    for (int l = 0; l < 3; ++l) { brW[br][l] = (const float*)d_in[p++]; brB[br][l] = (const float*)d_in[p++]; }
  const float *fuW[3], *fuB[3];
  for (int l = 0; l < 3; ++l) { fuW[l] = (const float*)d_in[p++]; fuB[l] = (const float*)d_in[p++]; }
  const float *ptW[3], *ptB[3];
  for (int l = 0; l < 3; ++l) { ptW[l] = (const float*)d_in[p++]; ptB[l] = (const float*)d_in[p++]; }
  const float *fsW[2], *fsB[2];
  for (int l = 0; l < 2; ++l) { fsW[l] = (const float*)d_in[p++]; fsB[l] = (const float*)d_in[p++]; }

  // workspace layout
  float* ws = (float*)d_ws;
  int*   knn_idx = (int*)ws;                                  // 4*4096*32 i32
  float* knn_d2  = ws + (long)Bn * Npts * KNN_K;              // 4*4096*32 f32
  float* net     = knn_d2 + (long)Bn * Npts * KNN_K;          // 4*64*4096 f32
  float* centers = net + (long)Bn * FD * Npts;                // 4*64*128 f32
  float* numb    = centers + (long)Bn * FD * NSPIX;           // 4*64*128 f32
  float* wsum    = numb + (long)Bn * FD * NSPIX;              // 4*128 f32
  _Float16* feats   = (_Float16*)(wsum + Bn * NSPIX);         // 4*192*4096 f16
  _Float16* tmpA    = feats + (long)Bn * 192 * Npts;          // 4*128*4096 f16
  _Float16* tmpB    = tmpA + (long)Bn * 128 * Npts;           // 4*128*4096 f16
  _Float16* concatb = tmpB + (long)Bn * 128 * Npts;           // 4*128*4096 f16

  // 1) KNN
  knn_kernel<<<Bn * Npts / 8, 256, 8 * Npts * sizeof(float), stream>>>(x, knn_idx, knn_d2);

  // 2) branch MLPs (WMMA)
  const float radii2[3] = {0.04f, 0.09f, 0.16f};
  size_t brShm = (size_t)(32 * 512 + 64 * 512 + 4 * 512 + 8 * 512) * 2 + 64 * 16 * 4;
  dim3 brGrid(Npts / 16, Bn);
  for (int br = 0; br < 3; ++br)
    branch_kernel<<<brGrid, 256, brShm, stream>>>(x, knn_idx, knn_d2,
        brW[br][0], brB[br][0], brW[br][1], brB[br][1], brW[br][2], brB[br][2],
        feats, radii2[br], br * 64);

  // 3) fuse / ptnet / fusion MLPs (WMMA GEMM chain, f16 activations)
  dim3 g(Npts / 128, Bn);
  auto shm = [](int Cin, int Cout) {
    int CinP = (Cin + 31) & ~31, CoutP = (Cout + 15) & ~15;
    return (size_t)(CoutP * CinP + CinP * 128 * 2) * 2;  // Wfrag + Xfrag + stage
  };
  // fuse: 192->128->64->64 (last into concat rows 64..127)
  mlp_gemm<<<g, 256, shm(192, 128), stream>>>(feats, 1, (long)192 * Npts, fuW[0], fuB[0],
                                              tmpA, 0, (long)128 * Npts, 192, 128, 1);
  mlp_gemm<<<g, 256, shm(128, 64), stream>>>(tmpA, 1, (long)128 * Npts, fuW[1], fuB[1],
                                             tmpB, 0, (long)64 * Npts, 128, 64, 1);
  mlp_gemm<<<g, 256, shm(64, 64), stream>>>(tmpB, 1, (long)64 * Npts, fuW[2], fuB[2],
                                            concatb + (long)64 * Npts, 0, (long)128 * Npts, 64, 64, 1);
  // ptnet: 3->64->128->64 (last into concat rows 0..63)
  mlp_gemm<<<g, 256, shm(3, 64), stream>>>(x, 0, (long)3 * Npts, ptW[0], ptB[0],
                                           tmpA, 0, (long)64 * Npts, 3, 64, 1);
  mlp_gemm<<<g, 256, shm(64, 128), stream>>>(tmpA, 1, (long)64 * Npts, ptW[1], ptB[1],
                                             tmpB, 0, (long)128 * Npts, 64, 128, 1);
  mlp_gemm<<<g, 256, shm(128, 64), stream>>>(tmpB, 1, (long)128 * Npts, ptW[2], ptB[2],
                                             concatb, 0, (long)128 * Npts, 128, 64, 1);
  // fusion: 128->64 (relu) -> 64 (no relu, f32) => net
  mlp_gemm<<<g, 256, shm(128, 64), stream>>>(concatb, 1, (long)128 * Npts, fsW[0], fsB[0],
                                             tmpA, 0, (long)64 * Npts, 128, 64, 1);
  mlp_gemm<<<g, 256, shm(64, 64), stream>>>(tmpA, 1, (long)64 * Npts, fsW[1], fsB[1],
                                            net, 1, (long)64 * Npts, 64, 64, 0);

  // 4) soft-SLIC
  center_init<<<(Bn * FD * NSPIX + 255) / 256, 256, 0, stream>>>(net, centers);
  size_t slShm = (size_t)(16 * 512 + 16 * 512) * 2                      // sCf, sPFf (f16)
               + (size_t)(128 + 8 * 16 * 128 + 64 * 128 + 128) * 4      // sCsq, sD, sNum, sW
               + (size_t)(128 * 16) * 4 * 2;                            // sIdx, sAff
  dim3 sg(Npts / 128, Bn);
  for (int it = 0; it < 10; ++it) {
    zero_kernel<<<64, 256, 0, stream>>>(numb, (long)Bn * FD * NSPIX + Bn * NSPIX);
    slic_assign<<<sg, 256, slShm, stream>>>(net, centers, numb, wsum, nullptr, nullptr, 0);
    center_update<<<(Bn * FD * NSPIX + 255) / 256, 256, 0, stream>>>(numb, wsum, centers);
  }

  // 5) final assignment + outputs: [Q | hard | centers | net]
  float* oQ    = (float*)d_out;
  float* oHard = oQ + (long)Bn * NSPIX * Npts;
  float* oCent = oHard + (long)Bn * Npts;
  float* oNet  = oCent + (long)Bn * FD * NSPIX;
  zero_kernel<<<512, 256, 0, stream>>>(oQ, (long)Bn * NSPIX * Npts);
  slic_assign<<<sg, 256, slShm, stream>>>(net, centers, numb, wsum, oQ, oHard, 1);
  copy_kernel<<<64, 256, 0, stream>>>(oCent, centers, (long)Bn * FD * NSPIX);
  copy_kernel<<<512, 256, 0, stream>>>(oNet, net, (long)Bn * FD * Npts);
}
